// SSD300_83245056131325
// MI455X (gfx1250) — matrix operand order repaired and checked by reading, not compiled, not run on hardware
//
#include <hip/hip_runtime.h>
#include <hip/hip_bf16.h>

#define BATCHN  4
#define NCLASS  21
#define NTOP    200
#define NPRIORS 8732

typedef __attribute__((ext_vector_type(16))) _Float16 v16h;
typedef __attribute__((ext_vector_type(8)))  _Float16 v8h;
typedef __attribute__((ext_vector_type(8)))  float    v8f;
union AV { v16h v; v8h h[2]; };

// ---------------- device constants (SSD prior geometry) ----------------
__device__ __constant__ int   c_offs[7] = {0, 5776, 7942, 8542, 8692, 8728, 8732};
__device__ __constant__ int   c_fmap[6] = {38, 19, 10, 5, 3, 1};
__device__ __constant__ int   c_nbox[6] = {4, 6, 6, 6, 4, 4};
__device__ __constant__ float c_step[6] = {8.f, 16.f, 32.f, 64.f, 100.f, 300.f};
__device__ __constant__ float c_smin[6] = {30.f, 60.f, 111.f, 162.f, 213.f, 264.f};
__device__ __constant__ float c_smax[6] = {60.f, 111.f, 162.f, 213.f, 264.f, 315.f};

// ---------------- one-time data transforms ----------------
// NCHW f32 image -> NHWC f16 with channels padded 3 -> 32 (zeros).
__global__ void cast_image(const float* __restrict__ x, _Float16* __restrict__ out, int HWin) {
    int idx = blockIdx.x * blockDim.x + threadIdx.x;
    int total = BATCHN * HWin * 32;
    if (idx >= total) return;
    int c   = idx & 31;
    int t   = idx >> 5;
    int pix = t % HWin;
    int b   = t / HWin;
    out[idx] = (c < 3) ? (_Float16)x[((size_t)b * 3 + c) * HWin + pix] : (_Float16)0.f;
}

// OIHW f32 weights -> [Cout][ky][kx][Cinp] f16 (Cin zero-padded to Cinp%32==0)
__global__ void cast_weight_nhwc(const float* __restrict__ w, _Float16* __restrict__ o,
                                 int Cin, int Cinp, int k, int total) {
    int idx = blockIdx.x * blockDim.x + threadIdx.x;
    if (idx >= total) return;
    int c  = idx % Cinp;
    int t  = idx / Cinp;
    int kx = t % k;  t /= k;
    int ky = t % k;
    int oc = t / k;
    o[idx] = (c < Cin) ? (_Float16)w[(((size_t)oc * Cin + c) * k + ky) * k + kx]
                       : (_Float16)0.f;
}

// ---------------- implicit-GEMM conv: 32x64 output tile per wave ----------------
// 8x v_wmma_f32_16x16x32_f16 per 32-wide K chunk. NHWC activations, K-order
// (ky,kx,cin) with Cinp%32==0 so A = two contiguous 16B loads per lane, B = one
// contiguous 32B load per lane. EXEC is uniform around every WMMA.
__global__ void __launch_bounds__(32)
conv_wmma(const _Float16* __restrict__ in, const _Float16* __restrict__ w,
          const float* __restrict__ bias, _Float16* __restrict__ out,
          int Cinp, int Hin, int Win, int Cout, int Hout, int Wout,
          int kw, int stride, int pad, int dil, int relu)
{
    const int lane   = threadIdx.x;
    const int laneHi = lane >> 4;
    const int lane16 = lane & 15;
    const int M     = Hout * Wout;
    const int Kpad  = kw * kw * Cinp;
    const int mbase = blockIdx.x * 32;
    const int nbase = blockIdx.y * 64;
    const int b     = blockIdx.z;
    const _Float16* inB = in + (size_t)b * Hin * Win * Cinp;

    int  iy0[2], ix0[2];
    bool mval[2];
#pragma unroll
    for (int mt = 0; mt < 2; ++mt) {
        int m = mbase + mt * 16 + lane16;
        mval[mt] = (m < M);
        int mm = mval[mt] ? m : 0;
        int oy = mm / Wout;
        int ox = mm - oy * Wout;
        iy0[mt] = oy * stride - pad;
        ix0[mt] = ox * stride - pad;
    }

    v8f acc[2][4];
#pragma unroll
    for (int mt = 0; mt < 2; ++mt)
#pragma unroll
        for (int nt = 0; nt < 4; ++nt) {
            v8f z = {};
            acc[mt][nt] = z;
        }
    v8h  zero8  = {};
    v16h zero16 = {};

    for (int ky = 0; ky < kw; ++ky) {
        for (int kx = 0; kx < kw; ++kx) {
            const _Float16* rp[2];
            bool rv[2];
#pragma unroll
            for (int mt = 0; mt < 2; ++mt) {
                int iy = iy0[mt] + ky * dil;
                int ix = ix0[mt] + kx * dil;
                rv[mt] = mval[mt] && iy >= 0 && iy < Hin && ix >= 0 && ix < Win;
                rp[mt] = inB + ((size_t)iy * Win + ix) * Cinp;
            }
            int koff = (ky * kw + kx) * Cinp;
            for (int cb = 0; cb < Cinp; cb += 32) {
                AV a[2];
#pragma unroll
                for (int mt = 0; mt < 2; ++mt) {
                    if (rv[mt]) {
                        const _Float16* p = rp[mt] + cb + 16 * laneHi;
                        a[mt].h[0] = *(const v8h*)p;          // K 16h+0..7
                        a[mt].h[1] = *(const v8h*)(p + 8);    // K 16h+8..15
                    } else {
                        a[mt].h[0] = zero8;
                        a[mt].h[1] = zero8;
                    }
                }
#pragma unroll
                for (int nt = 0; nt < 4; ++nt) {
                    int n = nbase + nt * 16 + lane16;
                    v16h bf = zero16;
                    if (n < Cout)
                        bf = *(const v16h*)(w + (size_t)n * Kpad + koff + cb + 16 * laneHi);
#pragma unroll
                    for (int mt = 0; mt < 2; ++mt)
                        acc[mt][nt] = __builtin_amdgcn_wmma_f32_16x16x32_f16(
                            false, a[mt].v, false, bf, (short)0, acc[mt][nt], false, false);
                }
            }
        }
    }

    // store NHWC: out[(b*M + m)*Cout + n]
#pragma unroll
    for (int nt = 0; nt < 4; ++nt) {
        int n = nbase + nt * 16 + lane16;
        if (n >= Cout) continue;
        float bv = bias[n];
#pragma unroll
        for (int mt = 0; mt < 2; ++mt) {
#pragma unroll
            for (int j = 0; j < 8; ++j) {
                int m = mbase + mt * 16 + j + 8 * laneHi;
                if (m < M) {
                    float vv = acc[mt][nt][j] + bv;
                    if (relu) vv = fmaxf(vv, 0.f);
                    out[((size_t)b * M + m) * Cout + n] = (_Float16)vv;
                }
            }
        }
    }
}

// ---------------- max pooling (NHWC, f16) ----------------
__global__ void maxpool_k(const _Float16* __restrict__ in, _Float16* __restrict__ out,
                          int C, int Hin, int Win, int Hout, int Wout,
                          int k, int s, int padL)
{
    int idx = blockIdx.x * blockDim.x + threadIdx.x;
    int total = BATCHN * C * Hout * Wout;
    if (idx >= total) return;
    int c  = idx % C;
    int t  = idx / C;
    int ox = t % Wout;  t /= Wout;
    int oy = t % Hout;
    int b  = t / Hout;
    const _Float16* ip = in + (size_t)b * Hin * Win * C;
    float mv = -1e30f;
    for (int ky = 0; ky < k; ++ky) {
        int iy = oy * s - padL + ky;
        if (iy < 0 || iy >= Hin) continue;
        for (int kx = 0; kx < k; ++kx) {
            int ix = ox * s - padL + kx;
            if (ix < 0 || ix >= Win) continue;
            mv = fmaxf(mv, (float)ip[((size_t)iy * Win + ix) * C + c]);
        }
    }
    out[idx] = (_Float16)mv;
}

// ---------------- channel L2 normalization (NHWC, conv4_3 source) ----------------
__global__ void __launch_bounds__(256)
l2norm_k(const _Float16* __restrict__ in, const float* __restrict__ gamma,
         _Float16* __restrict__ out, int C, int totalPix)
{
    int p = blockIdx.x * blockDim.x + threadIdx.x;
    if (p >= totalPix) return;
    const _Float16* ip = in  + (size_t)p * C;
    _Float16*       op = out + (size_t)p * C;
    float s = 0.f;
    for (int c = 0; c < C; ++c) {
        float v = (float)ip[c];
        s += v * v;
    }
    float scale = rsqrtf(s + 1e-10f);
    for (int c = 0; c < C; ++c)
        op[c] = (_Float16)((float)ip[c] * gamma[c] * scale);
}

// ---------------- head gather + prior gen + box decode + softmax ----------------
struct HeadPtrs { const _Float16* loc[6]; const _Float16* conf[6]; };

__global__ void __launch_bounds__(256)
decode_heads(HeadPtrs hp, float* __restrict__ boxes, float* __restrict__ conf)
{
    int idx = blockIdx.x * blockDim.x + threadIdx.x;
    if (idx >= BATCHN * NPRIORS) return;
    int b = idx / NPRIORS;
    int p = idx - b * NPRIORS;
    int s = 0;
    while (s < 5 && p >= c_offs[s + 1]) ++s;
    int q    = p - c_offs[s];
    int nb   = c_nbox[s];
    int f    = c_fmap[s];
    int cell = q / nb;
    int a    = q - cell * nb;
    int iy   = cell / f;
    int jx   = cell - iy * f;

    float sk  = c_smin[s] * (1.0f / 300.0f);
    float skp = sqrtf(sk * c_smax[s] * (1.0f / 300.0f));
    float pw, ph;
    if (a == 0)      { pw = sk;  ph = sk;  }
    else if (a == 1) { pw = skp; ph = skp; }
    else {
        float r = sqrtf(((a - 2) >> 1) ? 3.0f : 2.0f);
        if (((a - 2) & 1) == 0) { pw = sk * r; ph = sk / r; }
        else                    { pw = sk / r; ph = sk * r; }
    }
    float pcx = (jx + 0.5f) * c_step[s] * (1.0f / 300.0f);
    float pcy = (iy + 0.5f) * c_step[s] * (1.0f / 300.0f);
    pcx = fminf(fmaxf(pcx, 0.f), 1.f);
    pcy = fminf(fmaxf(pcy, 0.f), 1.f);
    pw  = fminf(fmaxf(pw , 0.f), 1.f);
    ph  = fminf(fmaxf(ph , 0.f), 1.f);

    // NHWC head buffers: channels contiguous per pixel
    size_t pix = (size_t)b * f * f + (size_t)iy * f + jx;
    const _Float16* lp = hp.loc[s] + pix * (4 * nb) + a * 4;
    float l0 = (float)lp[0];
    float l1 = (float)lp[1];
    float l2 = (float)lp[2];
    float l3 = (float)lp[3];
    float cx = pcx + l0 * 0.1f * pw;
    float cy = pcy + l1 * 0.1f * ph;
    float w  = pw * expf(l2 * 0.2f);
    float h  = ph * expf(l3 * 0.2f);
    float* bo = boxes + (size_t)idx * 4;
    bo[0] = cx - 0.5f * w; bo[1] = cy - 0.5f * h;
    bo[2] = cx + 0.5f * w; bo[3] = cy + 0.5f * h;

    const _Float16* cp = hp.conf[s] + pix * (NCLASS * nb) + a * NCLASS;
    float cv[NCLASS], mx = -1e30f;
    for (int j = 0; j < NCLASS; ++j) {
        cv[j] = (float)cp[j];
        mx = fmaxf(mx, cv[j]);
    }
    float sum = 0.f;
    for (int j = 0; j < NCLASS; ++j) { cv[j] = expf(cv[j] - mx); sum += cv[j]; }
    float inv = 1.0f / sum;
    float* co = conf + (size_t)idx * NCLASS;
    for (int j = 0; j < NCLASS; ++j) co[j] = cv[j] * inv;
}

// ---------------- per-(image,class) top-200 + greedy NMS ----------------
__global__ void __launch_bounds__(256)
nms_kernel(const float* __restrict__ boxes, const float* __restrict__ conf,
           float* __restrict__ out)
{
    int cls = blockIdx.x;
    int b   = blockIdx.y;
    int tid = threadIdx.x;
    float* o = out + ((size_t)(b * NCLASS + cls) * NTOP) * 5;
    if (cls == 0) {   // background class left empty
        for (int i = tid; i < NTOP * 5; i += 256) o[i] = 0.f;
        return;
    }
    __shared__ float s_sc[NPRIORS];
    __shared__ float t_sc[NTOP];
    __shared__ float t_bx[NTOP][4];
    __shared__ float t_area[NTOP];
    __shared__ float s_rem[NTOP];
    __shared__ float red_v[256];
    __shared__ int   red_i[256];

    for (int p = tid; p < NPRIORS; p += 256) {
        float s = conf[((size_t)b * NPRIORS + p) * NCLASS + cls];
        s_sc[p] = (s > 0.01f) ? s : 0.f;
    }
    __syncthreads();

    // --- top-200 (stable: ties pick lowest index, like lax.top_k) ---
    for (int it = 0; it < NTOP; ++it) {
        float bv = -1.f; int bi = 0;
        for (int p = tid; p < NPRIORS; p += 256) {
            float v = s_sc[p];
            if (v > bv) { bv = v; bi = p; }
        }
        red_v[tid] = bv; red_i[tid] = bi;
        __syncthreads();
        for (int off = 128; off > 0; off >>= 1) {
            if (tid < off) {
                if (red_v[tid + off] > red_v[tid] ||
                    (red_v[tid + off] == red_v[tid] && red_i[tid + off] < red_i[tid])) {
                    red_v[tid] = red_v[tid + off];
                    red_i[tid] = red_i[tid + off];
                }
            }
            __syncthreads();
        }
        if (tid == 0) {
            int j = red_i[0];
            t_sc[it] = (red_v[0] > 0.f) ? red_v[0] : 0.f;
            const float* bp = boxes + ((size_t)b * NPRIORS + j) * 4;
            t_bx[it][0] = bp[0]; t_bx[it][1] = bp[1];
            t_bx[it][2] = bp[2]; t_bx[it][3] = bp[3];
            t_area[it]  = (bp[2] - bp[0]) * (bp[3] - bp[1]);
            s_sc[j] = -1.f;
        }
        __syncthreads();
    }

    for (int i = tid; i < NTOP; i += 256) s_rem[i] = t_sc[i];
    __syncthreads();

    // --- greedy suppression over the 200 candidates ---
    for (int it = 0; it < NTOP; ++it) {
        float bv = -1.f; int bi = 0;
        if (tid < NTOP) { bv = s_rem[tid]; bi = tid; }
        red_v[tid] = bv; red_i[tid] = bi;
        __syncthreads();
        for (int off = 128; off > 0; off >>= 1) {
            if (tid < off) {
                if (red_v[tid + off] > red_v[tid] ||
                    (red_v[tid + off] == red_v[tid] && red_i[tid + off] < red_i[tid])) {
                    red_v[tid] = red_v[tid + off];
                    red_i[tid] = red_i[tid + off];
                }
            }
            __syncthreads();
        }
        int   j  = red_i[0];
        float sc = red_v[0];
        if (tid == 0) {
            if (sc > 0.f) {
                o[it * 5 + 0] = sc;
                o[it * 5 + 1] = t_bx[j][0]; o[it * 5 + 2] = t_bx[j][1];
                o[it * 5 + 3] = t_bx[j][2]; o[it * 5 + 4] = t_bx[j][3];
            } else {
                o[it * 5 + 0] = 0.f; o[it * 5 + 1] = 0.f; o[it * 5 + 2] = 0.f;
                o[it * 5 + 3] = 0.f; o[it * 5 + 4] = 0.f;
            }
        }
        if (tid < NTOP) {
            float lt0 = fmaxf(t_bx[j][0], t_bx[tid][0]);
            float lt1 = fmaxf(t_bx[j][1], t_bx[tid][1]);
            float rb0 = fminf(t_bx[j][2], t_bx[tid][2]);
            float rb1 = fminf(t_bx[j][3], t_bx[tid][3]);
            float ww  = fmaxf(rb0 - lt0, 0.f);
            float hh  = fmaxf(rb1 - lt1, 0.f);
            float inter = ww * hh;
            float iou = inter / (t_area[j] + t_area[tid] - inter + 1e-9f);
            if (iou > 0.45f) s_rem[tid] = 0.f;
            if (tid == j)    s_rem[tid] = 0.f;
        }
        __syncthreads();
    }
}

// =====================================================================
extern "C" void kernel_launch(void* const* d_in, const int* in_sizes, int n_in,
                              void* d_out, int out_size, void* d_ws, size_t ws_size,
                              hipStream_t stream)
{
    (void)in_sizes; (void)n_in; (void)out_size; (void)ws_size;
    struct Spec { int cin, cout, k; };
    static const Spec SPECS[35] = {
        {3,64,3},{64,64,3},{64,128,3},{128,128,3},
        {128,256,3},{256,256,3},{256,256,3},
        {256,512,3},{512,512,3},{512,512,3},
        {512,512,3},{512,512,3},{512,512,3},
        {512,1024,3},{1024,1024,1},
        {1024,256,1},{256,512,3},
        {512,128,1},{128,256,3},
        {256,128,1},{128,256,3},
        {256,128,1},{128,256,3},
        // heads interleaved: loc_i, conf_i
        {512,16,3},{512,84,3},
        {1024,24,3},{1024,126,3},
        {512,24,3},{512,126,3},
        {256,24,3},{256,126,3},
        {256,16,3},{256,84,3},
        {256,16,3},{256,84,3},
    };
    static const int FM[6] = {38,19,10,5,3,1};
    static const int NB[6] = {4,6,6,6,4,4};

    // ---- workspace carve-out (256B aligned) ----
    char* base = (char*)d_ws;
    size_t off = 0;
    auto alloc = [&](size_t bytes) -> char* {
        off = (off + 255) & ~(size_t)255;
        char* p = base + off;
        off += bytes;
        return p;
    };
    _Float16* actA = (_Float16*)alloc((size_t)BATCHN * 300 * 300 * 64 * 2);
    _Float16* actB = (_Float16*)alloc((size_t)BATCHN * 300 * 300 * 64 * 2);
    _Float16* w16[35]; int Cinp[35];
    for (int i = 0; i < 35; ++i) {
        Cinp[i] = (i == 0) ? 32 : SPECS[i].cin;        // pad image channels 3 -> 32
        w16[i] = (_Float16*)alloc((size_t)SPECS[i].cout * SPECS[i].k * SPECS[i].k * Cinp[i] * 2);
    }
    _Float16* src1 = (_Float16*)alloc((size_t)BATCHN * 38 * 38 * 512 * 2);
    _Float16 *locb[6], *confb[6];
    for (int s = 0; s < 6; ++s) {
        locb[s]  = (_Float16*)alloc((size_t)BATCHN * FM[s] * FM[s] * 4 * NB[s] * 2);
        confb[s] = (_Float16*)alloc((size_t)BATCHN * FM[s] * FM[s] * NCLASS * NB[s] * 2);
    }
    float* boxes = (float*)alloc((size_t)BATCHN * NPRIORS * 4 * sizeof(float));
    float* confp = (float*)alloc((size_t)BATCHN * NPRIORS * NCLASS * sizeof(float));

    // ---- one-time transforms: weights -> f16 HWI-padded, image -> NHWC f16 ----
    for (int i = 0; i < 35; ++i) {
        int tot = SPECS[i].cout * SPECS[i].k * SPECS[i].k * Cinp[i];
        cast_weight_nhwc<<<(tot + 255) / 256, 256, 0, stream>>>(
            (const float*)d_in[1 + 2 * i], w16[i], SPECS[i].cin, Cinp[i], SPECS[i].k, tot);
    }
    {
        int tot = BATCHN * 300 * 300 * 32;
        cast_image<<<(tot + 255) / 256, 256, 0, stream>>>((const float*)d_in[0], actA, 300 * 300);
    }

    int H = 300, W = 300;
    auto conv = [&](const _Float16* in, _Float16* out, int ci,
                    int stride, int pad, int dil, int relu) {
        const Spec& sp = SPECS[ci];
        int Ho = (H + 2 * pad - dil * (sp.k - 1) - 1) / stride + 1;
        int Wo = (W + 2 * pad - dil * (sp.k - 1) - 1) / stride + 1;
        int M  = Ho * Wo;
        dim3 g((M + 31) / 32, (sp.cout + 63) / 64, BATCHN);
        conv_wmma<<<g, 32, 0, stream>>>(in, w16[ci], (const float*)d_in[2 + 2 * ci], out,
                                        Cinp[ci], H, W, sp.cout, Ho, Wo,
                                        sp.k, stride, pad, dil, relu);
        H = Ho; W = Wo;
    };
    auto headconv = [&](const _Float16* in, _Float16* out, int ci) {
        const Spec& sp = SPECS[ci];
        int M = H * W;
        dim3 g((M + 31) / 32, (sp.cout + 63) / 64, BATCHN);
        conv_wmma<<<g, 32, 0, stream>>>(in, w16[ci], (const float*)d_in[2 + 2 * ci], out,
                                        Cinp[ci], H, W, sp.cout, H, W,
                                        sp.k, 1, 1, 1, 0);
    };
    auto pool = [&](const _Float16* in, _Float16* out, int C,
                    int k, int s, int padL, int padH) {
        int Ho = (H + padL + padH - k) / s + 1;
        int Wo = (W + padL + padH - k) / s + 1;
        int tot = BATCHN * C * Ho * Wo;
        maxpool_k<<<(tot + 255) / 256, 256, 0, stream>>>(in, out, C, H, W, Ho, Wo, k, s, padL);
        H = Ho; W = Wo;
    };

    // ---- VGG backbone + extras (ping-pong actA/actB, all NHWC f16) ----
    conv(actA, actB, 0, 1, 1, 1, 1);
    conv(actB, actA, 1, 1, 1, 1, 1);
    pool(actA, actB, 64, 2, 2, 0, 0);        // 150
    conv(actB, actA, 2, 1, 1, 1, 1);
    conv(actA, actB, 3, 1, 1, 1, 1);
    pool(actB, actA, 128, 2, 2, 0, 0);       // 75
    conv(actA, actB, 4, 1, 1, 1, 1);
    conv(actB, actA, 5, 1, 1, 1, 1);
    conv(actA, actB, 6, 1, 1, 1, 1);
    pool(actB, actA, 256, 2, 2, 0, 1);       // ceil -> 38
    conv(actA, actB, 7, 1, 1, 1, 1);
    conv(actB, actA, 8, 1, 1, 1, 1);
    conv(actA, actB, 9, 1, 1, 1, 1);         // conv4_3 in actB (38x38x512)
    {
        int totPix = BATCHN * H * W;
        l2norm_k<<<(totPix + 255) / 256, 256, 0, stream>>>(
            actB, (const float*)d_in[71], src1, 512, totPix);
        headconv(src1, locb[0], 23);
        headconv(src1, confb[0], 24);
    }
    pool(actB, actA, 512, 2, 2, 0, 0);       // 19
    conv(actA, actB, 10, 1, 1, 1, 1);
    conv(actB, actA, 11, 1, 1, 1, 1);
    conv(actA, actB, 12, 1, 1, 1, 1);
    pool(actB, actA, 512, 3, 1, 1, 1);       // 19
    conv(actA, actB, 13, 1, 6, 6, 1);        // conv6 dilated
    conv(actB, actA, 14, 1, 0, 1, 1);        // conv7 -> src2 (actA, 19)
    headconv(actA, locb[1], 25);
    headconv(actA, confb[1], 26);
    conv(actA, actB, 15, 1, 0, 1, 1);        // conv8_1
    conv(actB, actA, 16, 2, 1, 1, 1);        // conv8_2 -> src3 (actA, 10)
    headconv(actA, locb[2], 27);
    headconv(actA, confb[2], 28);
    conv(actA, actB, 17, 1, 0, 1, 1);        // conv9_1
    conv(actB, actA, 18, 2, 1, 1, 1);        // conv9_2 -> src4 (actA, 5)
    headconv(actA, locb[3], 29);
    headconv(actA, confb[3], 30);
    conv(actA, actB, 19, 1, 0, 1, 1);        // conv10_1
    conv(actB, actA, 20, 1, 0, 1, 1);        // conv10_2 -> src5 (actA, 3)
    headconv(actA, locb[4], 31);
    headconv(actA, confb[4], 32);
    conv(actA, actB, 21, 1, 0, 1, 1);        // conv11_1
    conv(actB, actA, 22, 1, 0, 1, 1);        // conv11_2 -> src6 (actA, 1)
    headconv(actA, locb[5], 33);
    headconv(actA, confb[5], 34);

    // ---- decode + softmax + NMS ----
    HeadPtrs hp;
    for (int s = 0; s < 6; ++s) { hp.loc[s] = locb[s]; hp.conf[s] = confb[s]; }
    int tot = BATCHN * NPRIORS;
    decode_heads<<<(tot + 255) / 256, 256, 0, stream>>>(hp, boxes, confp);
    dim3 gn(NCLASS, BATCHN);
    nms_kernel<<<gn, 256, 0, stream>>>(boxes, confp, (float*)d_out);
}
